// GraphSAGELayer_37203006718144
// MI455X (gfx1250) — matrix operand order repaired and checked by reading, not compiled, not run on hardware
//
#include <hip/hip_runtime.h>

// ---------------------------------------------------------------------------
// GraphSAGE layer for MI455X (gfx1250, wave32, WMMA).
//
//   agg = scatter_add(feat[src], dst)            (memset + kernel 1, L2-resident)
//   out = feat@W1^T + (agg/deg)@W2^T + b1 + b2   (kernel 2, fused K=1024 GEMM)
//
// GEMM uses split-bf16: x = hi + lo (bf16 planes built ONCE at LDS staging),
//   acc += Ahi*Bhi + Ahi*Blo + Alo*Bhi      via v_wmma_f32_16x16x32_bf16
// so the compute loop is ds_load_b128 + WMMA only (no per-fragment cvt VALU).
//
// Scratch: d_ws must hold N * 512 floats (102.4 MB) for `agg`.
// ---------------------------------------------------------------------------

typedef __attribute__((ext_vector_type(16))) __bf16        v16bf;
typedef __attribute__((ext_vector_type(8)))  float         v8f;
typedef __attribute__((ext_vector_type(4)))  unsigned int  u32x4;

#define F_IN   512
#define F_OUT  512
#define KTOT   1024          // feat(512) ++ agg/deg(512)
#define TM     128           // rows per workgroup
#define TN     128           // cols per workgroup
#define TK     32            // K per step == one bf16 WMMA K
#define NSTEPS (KTOT / TK)   // 32

union FragU {
    v16bf v;
    u32x4 u[2];
};

__device__ __forceinline__ unsigned short f32_to_bf16_rne(float x) {
    unsigned int u = __float_as_uint(x);
    u += 0x7FFFu + ((u >> 16) & 1u);
    return (unsigned short)(u >> 16);
}
__device__ __forceinline__ float bf16_up_bits(unsigned short h) {
    return __uint_as_float(((unsigned int)h) << 16);
}
// split 4 consecutive f32 into packed bf16 hi (RNE) and bf16 lo (trunc) pairs
__device__ __forceinline__ void split4(const float* x, uint2& hp, uint2& lp) {
    unsigned int h[4], l[4];
#pragma unroll
    for (int j = 0; j < 4; ++j) {
        unsigned int hj = f32_to_bf16_rne(x[j]);
        float res = x[j] - bf16_up_bits((unsigned short)hj);
        h[j] = hj;
        l[j] = __float_as_uint(res) >> 16;      // truncated residual
    }
    hp.x = h[0] | (h[1] << 16);  hp.y = h[2] | (h[3] << 16);
    lp.x = l[0] | (l[1] << 16);  lp.y = l[2] | (l[3] << 16);
}

// ---------------------------------------------------------------------------
// Kernel 1: edge scatter-add. One wave32 per edge; float4 gathers of the src
// row (feat = 102 MB, L2-resident), non-returning relaxed agent-scope f32
// atomics -> global_atomic_add_f32.
// ---------------------------------------------------------------------------
__global__ __launch_bounds__(256)
void sage_scatter_add(const float* __restrict__ feat,
                      const int*   __restrict__ src,
                      const int*   __restrict__ dst,
                      float*       __restrict__ agg,
                      int nEdges) {
    int e    = (blockIdx.x * blockDim.x + threadIdx.x) >> 5;
    int lane = threadIdx.x & 31;
    if (e >= nEdges) return;
    int s = src[e];
    int d = dst[e];
    const float4* fp = (const float4*)(feat + (size_t)s * F_IN);
    float*        ap = agg + (size_t)d * F_IN;
#pragma unroll
    for (int i = 0; i < F_IN / 128; ++i) {
        float4 v = fp[lane + i * 32];
        int base = (lane + i * 32) * 4;
        __hip_atomic_fetch_add(ap + base + 0, v.x, __ATOMIC_RELAXED, __HIP_MEMORY_SCOPE_AGENT);
        __hip_atomic_fetch_add(ap + base + 1, v.y, __ATOMIC_RELAXED, __HIP_MEMORY_SCOPE_AGENT);
        __hip_atomic_fetch_add(ap + base + 2, v.z, __ATOMIC_RELAXED, __HIP_MEMORY_SCOPE_AGENT);
        __hip_atomic_fetch_add(ap + base + 3, v.w, __ATOMIC_RELAXED, __HIP_MEMORY_SCOPE_AGENT);
    }
}

// ---------------------------------------------------------------------------
// Kernel 2: fused GEMM. 128x128 tile / 256-thread WG, 8 waves (2M x 4N),
// 64x32 per wave = 4x2 fragments, 3 WMMAs each. bf16 hi/lo planes in LDS,
// double buffered (4 planes x 2 x 128 x 32 x 2B = 64 KB of 320 KB WGP LDS).
// ---------------------------------------------------------------------------
__global__ __launch_bounds__(256)
void sage_fused_gemm(const float* __restrict__ feat,
                     const float* __restrict__ agg,
                     const float* __restrict__ in_deg,
                     const float* __restrict__ W1,
                     const float* __restrict__ b1,
                     const float* __restrict__ W2,
                     const float* __restrict__ b2,
                     float*       __restrict__ out,
                     int nRows) {
    __shared__ __align__(16) unsigned short AsH[2][TM][TK];  // 16 KB
    __shared__ __align__(16) unsigned short AsL[2][TM][TK];  // 16 KB
    __shared__ __align__(16) unsigned short BsH[2][TN][TK];  // 16 KB ([o][k])
    __shared__ __align__(16) unsigned short BsL[2][TN][TK];  // 16 KB

    const int tid     = threadIdx.x;
    const int lane    = tid & 31;
    const int wid     = tid >> 5;        // 0..7
    const int wm      = wid & 1;         // wave row (2)
    const int wn      = wid >> 1;        // wave col (4)
    const int rowBase = blockIdx.x * TM;
    const int colBase = blockIdx.y * TN;

    v8f acc[4][2];
    v8f zero = {};
#pragma unroll
    for (int mt = 0; mt < 4; ++mt)
#pragma unroll
        for (int nt = 0; nt < 2; ++nt) acc[mt][nt] = zero;

    float regA[16], regB[16];

    // ---- global -> registers (A': feat|agg/deg rows, B': W1|W2 rows) ----
    auto loadRegs = [&](int kt) {
        const int k0 = kt * TK;
#pragma unroll
        for (int i = 0; i < 4; ++i) {
            int c  = tid + i * 256;        // 0..1023
            int r  = c >> 3;               // tile row 0..127
            int kc = (c & 7) * 4;          // k chunk 0..28
            int kg = k0 + kc;
            // A tile
            float4 va = make_float4(0.f, 0.f, 0.f, 0.f);
            int g = rowBase + r;
            if (g < nRows) {
                if (kg < F_IN) {
                    va = *(const float4*)(feat + (size_t)g * F_IN + kg);
                } else {
                    va = *(const float4*)(agg + (size_t)g * F_IN + (kg - F_IN));
                    float dg = in_deg[g];
                    va.x /= dg; va.y /= dg; va.z /= dg; va.w /= dg;
                }
                // prefetch next K tile of the A stream (feat+agg > L2 capacity)
                if (i == 0 && kg + TK < KTOT) {
                    const float* pf = (kg + TK < F_IN)
                        ? feat + (size_t)g * F_IN + (kg + TK)
                        : agg  + (size_t)g * F_IN + (kg + TK - F_IN);
                    __builtin_prefetch(pf, 0, 1);
                }
            }
            regA[i * 4 + 0] = va.x; regA[i * 4 + 1] = va.y;
            regA[i * 4 + 2] = va.z; regA[i * 4 + 3] = va.w;
            // B tile (o = colBase + r is always < 512)
            int o = colBase + r;
            const float* wp = (kg < F_IN) ? (W1 + (size_t)o * F_IN + kg)
                                          : (W2 + (size_t)o * F_IN + (kg - F_IN));
            float4 vb = *(const float4*)wp;
            regB[i * 4 + 0] = vb.x; regB[i * 4 + 1] = vb.y;
            regB[i * 4 + 2] = vb.z; regB[i * 4 + 3] = vb.w;
        }
    };

    // ---- registers -> LDS bf16 hi/lo planes (split done ONCE per tile) ----
    auto stageLds = [&](int buf) {
#pragma unroll
        for (int i = 0; i < 4; ++i) {
            int c  = tid + i * 256;
            int r  = c >> 3;
            int kc = (c & 7) * 4;
            uint2 hp, lp;
            split4(&regA[i * 4], hp, lp);
            *(uint2*)&AsH[buf][r][kc] = hp;
            *(uint2*)&AsL[buf][r][kc] = lp;
            split4(&regB[i * 4], hp, lp);
            *(uint2*)&BsH[buf][r][kc] = hp;
            *(uint2*)&BsL[buf][r][kc] = lp;
        }
    };

    // ---- LDS -> fragments: pure b128 loads (contiguous k order in A and B) ----
    // A 16x32: lanes 0-15 hold M=lane, K in 0..15; lanes 16-31 K in 16..31.
    auto ldA = [&](const unsigned short* plane, int mt) -> v16bf {
        int ar = wm * 64 + mt * 16 + (lane & 15);
        int kb = (lane >> 4) * 16;
        const unsigned short* p = plane + ar * TK;
        FragU f;
        f.u[0] = *(const u32x4*)(p + kb);
        f.u[1] = *(const u32x4*)(p + kb + 8);
        return f.v;
    };
    // B 32x16: lane holds column o = lane&15, K = (lane>>4)*16 .. +15.
    auto ldB = [&](const unsigned short* plane, int nt) -> v16bf {
        int br = wn * 32 + nt * 16 + (lane & 15);
        int kb = (lane >> 4) * 16;
        const unsigned short* p = plane + br * TK + kb;
        FragU f;
        f.u[0] = *(const u32x4*)(p);
        f.u[1] = *(const u32x4*)(p + 8);
        return f.v;
    };

    auto compute = [&](int buf) {
        const unsigned short* pAH = &AsH[buf][0][0];
        const unsigned short* pAL = &AsL[buf][0][0];
        const unsigned short* pBH = &BsH[buf][0][0];
        const unsigned short* pBL = &BsL[buf][0][0];
        v16bf aH[4], aL[4];
#pragma unroll
        for (int mt = 0; mt < 4; ++mt) {
            aH[mt] = ldA(pAH, mt);
            aL[mt] = ldA(pAL, mt);
        }
#pragma unroll
        for (int nt = 0; nt < 2; ++nt) {
            v16bf bH = ldB(pBH, nt);
            v16bf bL = ldB(pBL, nt);
#pragma unroll
            for (int mt = 0; mt < 4; ++mt) {
                acc[mt][nt] = __builtin_amdgcn_wmma_f32_16x16x32_bf16(
                    false, aH[mt], false, bH, (short)0, acc[mt][nt], false, false);
                acc[mt][nt] = __builtin_amdgcn_wmma_f32_16x16x32_bf16(
                    false, aH[mt], false, bL, (short)0, acc[mt][nt], false, false);
                acc[mt][nt] = __builtin_amdgcn_wmma_f32_16x16x32_bf16(
                    false, aL[mt], false, bH, (short)0, acc[mt][nt], false, false);
            }
        }
    };

    // ---- main K loop, double buffered ----
    loadRegs(0);
    stageLds(0);
    __syncthreads();
    for (int kt = 0; kt < NSTEPS; ++kt) {
        int buf = kt & 1;
        if (kt + 1 < NSTEPS) loadRegs(kt + 1);   // overlap global latency
        compute(buf);
        __syncthreads();
        if (kt + 1 < NSTEPS) {
            stageLds(buf ^ 1);
            __syncthreads();
        }
    }

    // ---- epilogue: bias add + store (C/D layout: VGPR j -> M row j / j+8) ----
#pragma unroll
    for (int mt = 0; mt < 4; ++mt) {
#pragma unroll
        for (int nt = 0; nt < 2; ++nt) {
            int gcol   = colBase + wn * 32 + nt * 16 + (lane & 15);
            float bias = b1[gcol] + b2[gcol];
            int rbase  = rowBase + wm * 64 + mt * 16 + ((lane >> 4) << 3);
#pragma unroll
            for (int j = 0; j < 8; ++j) {
                int grow = rbase + j;
                if (grow < nRows)
                    out[(size_t)grow * F_OUT + gcol] = acc[mt][nt][j] + bias;
            }
        }
    }
}

// ---------------------------------------------------------------------------
extern "C" void kernel_launch(void* const* d_in, const int* in_sizes, int n_in,
                              void* d_out, int out_size, void* d_ws, size_t ws_size,
                              hipStream_t stream) {
    const float* feat = (const float*)d_in[0];
    const int*   src  = (const int*)  d_in[1];
    const int*   dst  = (const int*)  d_in[2];
    const float* deg  = (const float*)d_in[3];
    const float* W1   = (const float*)d_in[4];
    const float* b1   = (const float*)d_in[5];
    const float* W2   = (const float*)d_in[6];
    const float* b2   = (const float*)d_in[7];
    float*       out  = (float*)d_out;

    const int nRows  = in_sizes[3];   // N (in_deg length)
    const int nEdges = in_sizes[1];   // E (src_idx length)

    // agg scratch: nRows * 512 f32 (102.4 MB), zeroed every call.
    float* agg = (float*)d_ws;
    hipMemsetAsync(agg, 0, (size_t)nRows * F_IN * sizeof(float), stream);

    {   // one wave32 per edge, 8 edges per 256-thread block
        int blocks = (nEdges + 7) / 8;
        sage_scatter_add<<<blocks, 256, 0, stream>>>(feat, src, dst, agg, nEdges);
    }
    {   // fused GEMM: grid = ceil(N/128) x (512/128)
        dim3 grid((nRows + TM - 1) / TM, F_OUT / TN);
        sage_fused_gemm<<<grid, 256, 0, stream>>>(feat, agg, deg, W1, b1, W2, b2,
                                                  out, nRows);
    }
}
